// WindowAttention_78520592105801
// MI455X (gfx1250) — matrix operand order repaired and checked by reading, not compiled, not run on hardware
//
#include <hip/hip_runtime.h>

// ---------------------------------------------------------------------------
// Window attention for MI455X (gfx1250), wave32 + v_wmma_f32_16x16x32_f16.
//
// Memory-bound problem (~84 GFLOP vs a few hundred MB) -> f16 operands / f32
// accum, fused window (un)partition, operand layouts chosen so every WMMA
// fragment is a plain contiguous 16B/32B vector load per lane, per the
// CDNA5 ISA 7.12.2 layouts:
//   A (16x32 f16): lane<16 row=lane, elems0-15=K0-15 contiguous;
//                  lane>=16 same row, K16-31 (this code's order, not the ISA table order).
//   B (32x16 f16), same code order: lane<16 col=lane holds K0-15 contiguous; lane>=16 K16-31.
//   C/D (16x16 f32): lane holds col = lane&15; VGPR i = row (lane>>4)*8 + i.
//
// Round-2 fixes: x is pre-converted to f16 in window-permuted order (one
// pass) so the QKV GEMM streams half the bytes and carries no cvt chain;
// V B-fragments hoisted in attention; global_prefetch_b8 for next K-chunk.
// ---------------------------------------------------------------------------

typedef _Float16 v16h __attribute__((ext_vector_type(16)));
typedef _Float16 v8h  __attribute__((ext_vector_type(8)));
typedef float    v8f  __attribute__((ext_vector_type(8)));

#define C_DIM   384
#define TC_DIM  1152          // 3*C
#define NH      12
#define HD      32
#define B_DIM   16
#define G_DIM   64            // windows per image
#define WT      64            // tokens per window
#define N_DIM   4096

__device__ __forceinline__ v8f wmma16(v16h a, v16h b, v8f c) {
    return __builtin_amdgcn_wmma_f32_16x16x32_f16(false, a, false, b,
                                                  (short)0, c, false, false);
}

__device__ __forceinline__ v16h combine8(v8h lo, v8h hi) {
    v16h a;
#pragma unroll
    for (int i = 0; i < 8; ++i) { a[i] = lo[i]; a[i + 8] = hi[i]; }
    return a;
}

// A-fragment (16x32) from row-major f16 [.][ld], tile row r, K offset k0.
__device__ __forceinline__ v16h load_a_f16(const _Float16* base, int ld,
                                           int r, int k0, int half) {
    const _Float16* p0 = base + (size_t)r * ld + k0 + half * 16;
    return combine8(*(const v8h*)p0, *(const v8h*)(p0 + 8));
}

// B-fragment (32x16): source is "n-major" Wt[n][k] so lane n reads 16
// contiguous K values (32 bytes).
__device__ __forceinline__ v16h load_b_f16(const _Float16* base, int ld,
                                           int n, int k0, int half) {
    return *(const v16h*)(base + (size_t)n * ld + k0 + half * 16);
}

// --------------------------- K0a: weight convert ---------------------------
// wt[n][k] = (f16) w[k][n]
__global__ void cvt_wt_kernel(const float* __restrict__ w,
                              _Float16* __restrict__ wt, int K, int Nout) {
    int idx = blockIdx.x * blockDim.x + threadIdx.x;
    if (idx >= K * Nout) return;
    int n = idx / K;
    int k = idx - n * K;
    wt[idx] = (_Float16)w[(size_t)k * Nout + n];
}

// --------------------------- K0b: x convert + window permute ---------------
// xg[(bg*64 + t)][c] = (f16) x[b][n(gy,gx,ty,tx)][c] : one pass, coalesced
// along c on both sides, so the QKV GEMM's A operand is contiguous rows.
__global__ void cvt_x_kernel(const float* __restrict__ x,
                             _Float16* __restrict__ xg) {
    int idx = blockIdx.x * blockDim.x + threadIdx.x;   // < 65536*384
    int row = idx / C_DIM;
    int c   = idx - row * C_DIM;
    int bg  = row >> 6, t = row & 63;
    int b   = bg >> 6,  g = bg & 63;
    int gy  = g >> 3,   gx = g & 7;
    int ty  = t >> 3,   tx = t & 7;
    int n   = (gy * 8 + ty) * 64 + (gx * 8 + tx);
    xg[idx] = (_Float16)x[((size_t)b * N_DIM + n) * C_DIM + c];
}

// --------------------------- K1: QKV projection ----------------------------
// One wave per (window bg, 64-col group). Computes [64 tok x 64 cols],
// K-loop over 384 channels.
__global__ void __launch_bounds__(32, 1)
qkv_kernel(const _Float16* __restrict__ xg,
           const _Float16* __restrict__ qkv_wt,
           _Float16* __restrict__ q_buf,   // [bg*nh][64][32]
           _Float16* __restrict__ k_buf,   // [bg*nh][64][32]
           _Float16* __restrict__ v_t) {   // [bg*nh][32][64]
    const int bg   = blockIdx.x;          // 0..1023
    const int ng   = blockIdx.y;          // 0..17
    const int lane = threadIdx.x;
    const int r    = lane & 15;
    const int half = lane >> 4;

    const _Float16* abase = xg + (size_t)bg * WT * C_DIM;
    const _Float16* bbase = qkv_wt + ((size_t)ng * 64 + r) * C_DIM + half * 16;

    v8f acc[4][4];
#pragma unroll
    for (int mi = 0; mi < 4; ++mi)
#pragma unroll
        for (int ni = 0; ni < 4; ++ni) acc[mi][ni] = (v8f)0.f;

    for (int kb = 0; kb < C_DIM; kb += 32) {
        // Prefetch next K-chunk (gfx1250 global_prefetch_b8 path).
        if (kb + 32 < C_DIM) {
            __builtin_prefetch(abase + (size_t)r * C_DIM + kb + 32, 0, 3);
            __builtin_prefetch(bbase + kb + 32, 0, 3);
        }
        v16h a[4];
#pragma unroll
        for (int mi = 0; mi < 4; ++mi)
            a[mi] = load_a_f16(abase, C_DIM, mi * 16 + r, kb, half);
        // One B-fragment live at a time -> low register pressure.
#pragma unroll
        for (int ni = 0; ni < 4; ++ni) {
            v16h bf = *(const v16h*)(bbase + (size_t)ni * 16 * C_DIM + kb);
#pragma unroll
            for (int mi = 0; mi < 4; ++mi)
                acc[mi][ni] = wmma16(a[mi], bf, acc[mi][ni]);
        }
    }

    const float scale = 0.17677669529663687f;  // hd^-0.5 folded into Q
#pragma unroll
    for (int ni = 0; ni < 4; ++ni) {
        int col   = ng * 64 + ni * 16 + r;
        int which = col / C_DIM;               // uniform per 16-col tile
        int hcol  = col - which * C_DIM;
        int head  = hcol >> 5;
        int d     = hcol & 31;
        size_t bh = (size_t)bg * NH + head;
#pragma unroll
        for (int mi = 0; mi < 4; ++mi) {
            int t0 = mi * 16 + half * 8;
            if (which == 2) {                  // V: store transposed, packed
                v8h pk;
#pragma unroll
                for (int i = 0; i < 8; ++i) pk[i] = (_Float16)acc[mi][ni][i];
                *(v8h*)(v_t + bh * (HD * WT) + (size_t)d * WT + t0) = pk;
            } else {
                _Float16* dst = (which == 0 ? q_buf : k_buf) +
                                bh * (WT * HD) + (size_t)t0 * HD + d;
                float s = (which == 0) ? scale : 1.0f;
#pragma unroll
                for (int i = 0; i < 8; ++i)
                    dst[(size_t)i * HD] = (_Float16)(acc[mi][ni][i] * s);
            }
        }
    }
}

// --------------------------- K2: attention ---------------------------------
// One wave per (window, head).  S^T = K @ Q^T so the softmax reduction over
// tokens runs along per-lane VGPRs (+1 shfl_xor(16)); P goes to LDS via
// 16B ds_store_b128, O = P @ V with A re-read from LDS.
#define P_LD 72   // f16 stride; 144B rows avoid systematic bank conflicts

__global__ void __launch_bounds__(32, 1)
attn_kernel(const _Float16* __restrict__ q_buf,
            const _Float16* __restrict__ k_buf,
            const _Float16* __restrict__ v_t,
            _Float16* __restrict__ o_all) {  // [65536][384]
    __shared__ _Float16 Pbuf[WT * P_LD];

    const int bg   = blockIdx.x;
    const int head = blockIdx.y;
    const size_t bh = (size_t)bg * NH + head;
    const int lane = threadIdx.x;
    const int r    = lane & 15;
    const int half = lane >> 4;

    const _Float16* qp = q_buf + bh * (WT * HD);
    const _Float16* kp = k_buf + bh * (WT * HD);
    const _Float16* vp = v_t   + bh * (HD * WT);

    // S^T[token][qrow]: A = K (4 token-tiles), B = Q^T (4 qrow-tiles), K=32.
    // Hold the 4 B-fragments; stream A one tile at a time.
    v16h bq[4];
#pragma unroll
    for (int ni = 0; ni < 4; ++ni) bq[ni] = load_b_f16(qp, HD, ni * 16 + r, 0, half);

    v8f s[4][4];
#pragma unroll
    for (int mi = 0; mi < 4; ++mi) {
        v16h ak = load_a_f16(kp, HD, mi * 16 + r, 0, half);
#pragma unroll
        for (int ni = 0; ni < 4; ++ni)
            s[mi][ni] = wmma16(ak, bq[ni], (v8f)0.f);
    }

    // Softmax over tokens (M direction) for each of the 4 qrow columns/lane.
#pragma unroll
    for (int ni = 0; ni < 4; ++ni) {
        float m = -3.0e38f;
#pragma unroll
        for (int mi = 0; mi < 4; ++mi)
#pragma unroll
            for (int i = 0; i < 8; ++i) m = fmaxf(m, s[mi][ni][i]);
        m = fmaxf(m, __shfl_xor(m, 16, 32));
        float sum = 0.f;
#pragma unroll
        for (int mi = 0; mi < 4; ++mi)
#pragma unroll
            for (int i = 0; i < 8; ++i) {
                float e = __expf(s[mi][ni][i] - m);
                s[mi][ni][i] = e;
                sum += e;
            }
        sum += __shfl_xor(sum, 16, 32);
        float rs = 1.0f / sum;
#pragma unroll
        for (int mi = 0; mi < 4; ++mi) {
            v8h pk;
#pragma unroll
            for (int i = 0; i < 8; ++i) pk[i] = (_Float16)(s[mi][ni][i] * rs);
            *(v8h*)&Pbuf[(ni * 16 + r) * P_LD + mi * 16 + half * 8] = pk;
        }
    }
    __syncthreads();

    // O = P @ V : A = P from LDS (qrow rows), B = V^T (hd cols), K = 64.
    v8f o[4][2];
#pragma unroll
    for (int mi = 0; mi < 4; ++mi)
#pragma unroll
        for (int nj = 0; nj < 2; ++nj) o[mi][nj] = (v8f)0.f;
#pragma unroll
    for (int kc = 0; kc < 2; ++kc) {
        v16h vb[2];
#pragma unroll
        for (int nj = 0; nj < 2; ++nj)
            vb[nj] = load_b_f16(vp, WT, nj * 16 + r, kc * 32, half);
#pragma unroll
        for (int mi = 0; mi < 4; ++mi) {
            const _Float16* p0 = &Pbuf[(mi * 16 + r) * P_LD + kc * 32 + half * 16];
            v16h pa = combine8(*(const v8h*)p0, *(const v8h*)(p0 + 8));
#pragma unroll
            for (int nj = 0; nj < 2; ++nj)
                o[mi][nj] = wmma16(pa, vb[nj], o[mi][nj]);
        }
    }

#pragma unroll
    for (int mi = 0; mi < 4; ++mi)
#pragma unroll
        for (int nj = 0; nj < 2; ++nj) {
            size_t row0 = (size_t)bg * WT + mi * 16 + half * 8;
            int col = head * HD + nj * 16 + r;
#pragma unroll
            for (int i = 0; i < 8; ++i)
                o_all[(row0 + i) * C_DIM + col] = (_Float16)o[mi][nj][i];
        }
}

// --------------------------- K3: output projection -------------------------
// Y = O @ proj_w + b, with reverse window partition fused into the store.
__global__ void __launch_bounds__(32, 1)
proj_kernel(const _Float16* __restrict__ o_all,
            const _Float16* __restrict__ proj_wt,
            const float* __restrict__ proj_b,
            float* __restrict__ out) {
    const int bg   = blockIdx.x;          // window / 64-row block
    const int ng   = blockIdx.y;          // 0..5 (64 cols each)
    const int lane = threadIdx.x;
    const int r    = lane & 15;
    const int half = lane >> 4;

    const _Float16* abase = o_all + (size_t)bg * WT * C_DIM;
    const _Float16* bbase = proj_wt + ((size_t)ng * 64 + r) * C_DIM + half * 16;

    v8f acc[4][4];
#pragma unroll
    for (int mi = 0; mi < 4; ++mi)
#pragma unroll
        for (int ni = 0; ni < 4; ++ni) acc[mi][ni] = (v8f)0.f;

    for (int kb = 0; kb < C_DIM; kb += 32) {
        if (kb + 32 < C_DIM) {
            __builtin_prefetch(abase + (size_t)r * C_DIM + kb + 32, 0, 3);
            __builtin_prefetch(bbase + kb + 32, 0, 3);
        }
        v16h a[4];
#pragma unroll
        for (int mi = 0; mi < 4; ++mi)
            a[mi] = load_a_f16(abase, C_DIM, mi * 16 + r, kb, half);
#pragma unroll
        for (int ni = 0; ni < 4; ++ni) {
            v16h bf = *(const v16h*)(bbase + (size_t)ni * 16 * C_DIM + kb);
#pragma unroll
            for (int mi = 0; mi < 4; ++mi)
                acc[mi][ni] = wmma16(a[mi], bf, acc[mi][ni]);
        }
    }

    const int b  = bg >> 6, g = bg & 63;
    const int gy = g >> 3,  gx = g & 7;
#pragma unroll
    for (int ni = 0; ni < 4; ++ni) {
        int col = ng * 64 + ni * 16 + r;
        float bias = proj_b[col];
#pragma unroll
        for (int mi = 0; mi < 4; ++mi) {
#pragma unroll
            for (int i = 0; i < 8; ++i) {
                int t  = mi * 16 + half * 8 + i;
                int ty = t >> 3, tx = t & 7;
                int n  = (gy * 8 + ty) * 64 + (gx * 8 + tx);
                out[((size_t)b * N_DIM + n) * C_DIM + col] = acc[mi][ni][i] + bias;
            }
        }
    }
}

// --------------------------- host launch -----------------------------------
extern "C" void kernel_launch(void* const* d_in, const int* in_sizes, int n_in,
                              void* d_out, int out_size, void* d_ws, size_t ws_size,
                              hipStream_t stream) {
    const float* x      = (const float*)d_in[0];
    const float* qkv_w  = (const float*)d_in[1];
    const float* proj_w = (const float*)d_in[2];
    const float* proj_b = (const float*)d_in[3];
    float* out = (float*)d_out;

    char* ws = (char*)d_ws;
    const size_t QKV_WT_BYTES  = (size_t)TC_DIM * C_DIM * 2;        //   884,736
    const size_t PROJ_WT_BYTES = (size_t)C_DIM * C_DIM * 2;         //   294,912
    const size_t QKV_BUF_BYTES = (size_t)B_DIM * G_DIM * NH * WT * HD * 2; // 50,331,648
    const size_t XG_BYTES      = (size_t)B_DIM * N_DIM * C_DIM * 2;        // 50,331,648

    _Float16* qkv_wt  = (_Float16*)(ws);
    _Float16* proj_wt = (_Float16*)(ws + QKV_WT_BYTES);
    _Float16* q_buf   = (_Float16*)(ws + QKV_WT_BYTES + PROJ_WT_BYTES);
    _Float16* k_buf   = (_Float16*)((char*)q_buf + QKV_BUF_BYTES);
    _Float16* v_t     = (_Float16*)((char*)k_buf + QKV_BUF_BYTES);
    _Float16* o_all   = (_Float16*)((char*)v_t  + QKV_BUF_BYTES);
    _Float16* xg      = (_Float16*)((char*)o_all + QKV_BUF_BYTES);

    cvt_wt_kernel<<<(TC_DIM * C_DIM + 255) / 256, 256, 0, stream>>>(
        qkv_w, qkv_wt, C_DIM, TC_DIM);
    cvt_wt_kernel<<<(C_DIM * C_DIM + 255) / 256, 256, 0, stream>>>(
        proj_w, proj_wt, C_DIM, C_DIM);
    cvt_x_kernel<<<(B_DIM * N_DIM * C_DIM) / 256, 256, 0, stream>>>(x, xg);

    qkv_kernel<<<dim3(B_DIM * G_DIM, TC_DIM / 64), 32, 0, stream>>>(
        xg, qkv_wt, q_buf, k_buf, v_t);

    attn_kernel<<<dim3(B_DIM * G_DIM, NH), 32, 0, stream>>>(
        q_buf, k_buf, v_t, o_all);

    proj_kernel<<<dim3(B_DIM * G_DIM, C_DIM / 64), 32, 0, stream>>>(
        o_all, proj_wt, proj_b, out);
}
